// LSTMStockPriceModel_47038481826534
// MI455X (gfx1250) — matrix operand order repaired and checked by reading, not compiled, not run on hardware
//
#include <hip/hip_runtime.h>

// ---------------------------------------------------------------------------
// Fused 2-layer LSTM (H=8) + linear head for MI455X (gfx1250, wave32, WMMA).
//
// One wave handles 16 batch rows for all T=512 steps; both layers fused so
// the intermediate sequence y0 is never written to memory. Per step per
// layer: gates[16g x 16b] = W[16x32] x state[32x16] + bias  via two
// v_wmma_f32_16x16x32_f16 (gate tiles {i,f} and {g,o}).
//
//   A (weights, time-invariant):  K0..7 = input vec, K8..15 = hidden, rest 0.
//     16-bit A 16x32 layout: VGPR v(0..3): lanes<16 -> K=2v,2v+1 (W_in),
//     VGPR 4+v: lanes<16 -> W_hh cols 2v,2v+1; lanes>=16 all 0.
//   B (state, rebuilt each step): K0..15 live in lanes 0..15 (2 f16/VGPR),
//     K16..31 in lanes 16..31 -> our pad zone. All real data stays in the
//     low half-wave, so B is built with v_cvt_pkrtz only (no shuffles).
//   C/D: VGPR j: lanes<16 = gate j, lanes>=16 = gate j+8 (batch = lane%16).
//     f/o gates are fetched from the high half with ds_swizzle xor16.
// ---------------------------------------------------------------------------

typedef __attribute__((ext_vector_type(16))) _Float16 v16h;
typedef __attribute__((ext_vector_type(8)))  float    v8f;

union AV { v16h v; unsigned int u[8]; };

__device__ __forceinline__ unsigned int pkh2(float a, float b) {
  auto p = __builtin_amdgcn_cvt_pkrtz(a, b);          // v_cvt_pkrtz_f16_f32
  return __builtin_bit_cast(unsigned int, p);
}

// lane <- lane ^ 16 (group-of-32 swizzle: and=0x1f, or=0, xor=0x10)
__device__ __forceinline__ float swz_x16(float x) {
  int r = __builtin_amdgcn_ds_swizzle(__builtin_bit_cast(int, x), 0x401f);
  return __builtin_bit_cast(float, r);
}

__device__ __forceinline__ float ftanh(float x) {
#if __has_builtin(__builtin_amdgcn_tanhf)
  return __builtin_amdgcn_tanhf(x);                   // v_tanh_f32 (1 trans op)
#else
  float t = __builtin_exp2f(x * 2.8853900817779268f); // 2*log2(e); v_exp_f32
  return 1.0f - 2.0f / (t + 1.0f);
#endif
}
__device__ __forceinline__ float fsigmoid(float x) {  // sigma(x)=0.5+0.5*tanh(x/2)
  return fmaf(0.5f, ftanh(0.5f * x), 0.5f);
}

__device__ __forceinline__ v8f wmma16(v16h a, v16h b, v8f c) {
  return __builtin_amdgcn_wmma_f32_16x16x32_f16(
      /*neg_a=*/false, a, /*neg_b=*/false, b,
      /*c_mod=*/(short)0, c, /*reuse_a=*/false, /*reuse_b=*/false);
}

// Pointwise LSTM cell update; valid in lanes 0..15 (batch = lane).
__device__ __forceinline__ void lstm_point(const v8f& d_if, const v8f& d_go,
                                           float (&h)[8], float (&c)[8]) {
#pragma unroll
  for (int j = 0; j < 8; ++j) {
    float si = fsigmoid(d_if[j]);            // gate i_j (low half)
    float sf = fsigmoid(swz_x16(d_if[j]));   // gate f_j lives in high half
    float tg = ftanh(d_go[j]);               // gate g_j (low half)
    float so = fsigmoid(swz_x16(d_go[j]));   // gate o_j lives in high half
    c[j] = fmaf(sf, c[j], si * tg);
    h[j] = so * ftanh(c[j]);
  }
}

__global__ __launch_bounds__(256) void lstm2_wmma_kernel(
    const float* __restrict__ x,
    const float* __restrict__ Wih0, const float* __restrict__ Whh0,
    const float* __restrict__ bih0, const float* __restrict__ bhh0,
    const float* __restrict__ Wih1, const float* __restrict__ Whh1,
    const float* __restrict__ bih1, const float* __restrict__ bhh1,
    const float* __restrict__ Wlin, const float* __restrict__ blin,
    float* __restrict__ out, int B, int T) {
  const int lane  = threadIdx.x & 31;
  const int waveg = (int)((blockIdx.x * blockDim.x + threadIdx.x) >> 5);
  const int base  = waveg * 16;
  if (base >= B) return;                       // whole-wave uniform exit

  const int  m    = lane & 15;                 // row within 16-wide tile
  const int  half = lane >> 4;
  const bool islo = (half == 0);

  // ---- A operands: weights, built once, constant across all 512 steps ----
  auto buildA = [&](const float* Wih, int icols, const float* Whh,
                    int tile) -> v16h {
    AV a;
    const int g = tile * 16 + m;               // global gate row 0..31
#pragma unroll
    for (int v = 0; v < 4; ++v) {
      const int k0 = 2 * v, k1 = 2 * v + 1;
      float w0 = 0.0f, w1 = 0.0f, u0 = 0.0f, u1 = 0.0f;
      if (half == 0) {                         // K = k0,k1 -> input, K = 8+k0 -> hidden
        w0 = (k0 < icols) ? Wih[g * icols + k0] : 0.0f;
        w1 = (k1 < icols) ? Wih[g * icols + k1] : 0.0f;
        u0 = Whh[g * 8 + k0];
        u1 = Whh[g * 8 + k1];
      }
      a.u[v] = pkh2(w0, w1);
      a.u[4 + v] = pkh2(u0, u1);
    }                                          // lanes>=16: K=16..31 zero pad
    return a.v;
  };
  const v16h A_if0 = buildA(Wih0, 1, Whh0, 0);
  const v16h A_go0 = buildA(Wih0, 1, Whh0, 1);
  const v16h A_if1 = buildA(Wih1, 8, Whh1, 0);
  const v16h A_go1 = buildA(Wih1, 8, Whh1, 1);

  // ---- C operands: bias replicated into the D layout (free bias add) ----
  auto buildC = [&](const float* bi, const float* bh, int tile) -> v8f {
    v8f cb;
#pragma unroll
    for (int j = 0; j < 8; ++j) {
      const int g = tile * 16 + j + 8 * half;  // lanes>=16 hold M=j+8
      cb[j] = bi[g] + bh[g];
    }
    return cb;
  };
  const v8f C_if0 = buildC(bih0, bhh0, 0);
  const v8f C_go0 = buildC(bih0, bhh0, 1);
  const v8f C_if1 = buildC(bih1, bhh1, 0);
  const v8f C_go1 = buildC(bih1, bhh1, 1);

  // ---- recurrent state: lanes 0..15 = batch rows, 8 regs per array ----
  float h0[8], c0[8], h1[8], c1[8];
#pragma unroll
  for (int j = 0; j < 8; ++j) { h0[j] = c0[j] = h1[j] = c1[j] = 0.0f; }

  // x is [B, T]: each lane streams its own contiguous row (float4 chunks).
  const float* xrow = x + (size_t)(base + m) * (size_t)T;

  for (int t4 = 0; t4 < T; t4 += 4) {
    const float4 xq = *(const float4*)(xrow + t4);
    const float xs[4] = {xq.x, xq.y, xq.z, xq.w};
#pragma unroll
    for (int k = 0; k < 4; ++k) {
      // ---------- layer 0: state = [x_t, 0..0 | h0] ----------
      AV b0;
      b0.u[0] = islo ? pkh2(xs[k], 0.0f) : 0u;  // K=0,1
      b0.u[1] = b0.u[2] = b0.u[3] = 0u;         // K=2..7 (D=1 pad)
#pragma unroll
      for (int v = 0; v < 4; ++v)               // K=8..15 = h0
        b0.u[4 + v] = islo ? pkh2(h0[2 * v], h0[2 * v + 1]) : 0u;
      v8f d_if = wmma16(A_if0, b0.v, C_if0);
      v8f d_go = wmma16(A_go0, b0.v, C_go0);
      lstm_point(d_if, d_go, h0, c0);

      // ---------- layer 1: state = [h0 | h1] ----------
      AV b1;
#pragma unroll
      for (int v = 0; v < 4; ++v) {
        b1.u[v]     = islo ? pkh2(h0[2 * v], h0[2 * v + 1]) : 0u;  // input=y0
        b1.u[4 + v] = islo ? pkh2(h1[2 * v], h1[2 * v + 1]) : 0u;  // hidden
      }
      d_if = wmma16(A_if1, b1.v, C_if1);
      d_go = wmma16(A_go1, b1.v, C_go1);
      lstm_point(d_if, d_go, h1, c1);
    }
  }

  // ---- epilogue: y = h1 @ Wlin.T + b ; emit y, h[2,B,8], c[2,B,8] ----
  if (islo) {
    const int n = base + lane;
    if (n < B) {
      float y = blin[0];
#pragma unroll
      for (int j = 0; j < 8; ++j) y = fmaf(Wlin[j], h1[j], y);
      const size_t Bz = (size_t)B;
      out[n] = y;
      float* ho0 = out + Bz;                 // h layer 0
      float* ho1 = out + Bz + 8 * Bz;        // h layer 1
      float* co0 = out + 17 * Bz;            // c layer 0
      float* co1 = out + 17 * Bz + 8 * Bz;   // c layer 1
      const size_t r = (size_t)n * 8;
#pragma unroll
      for (int j = 0; j < 8; ++j) {
        ho0[r + j] = h0[j];
        ho1[r + j] = h1[j];
        co0[r + j] = c0[j];
        co1[r + j] = c1[j];
      }
    }
  }
}

extern "C" void kernel_launch(void* const* d_in, const int* in_sizes, int n_in,
                              void* d_out, int out_size, void* d_ws,
                              size_t ws_size, hipStream_t stream) {
  (void)n_in; (void)d_ws; (void)ws_size;
  const float* x    = (const float*)d_in[0];
  const float* Wih0 = (const float*)d_in[1];
  const float* Whh0 = (const float*)d_in[2];
  const float* bih0 = (const float*)d_in[3];
  const float* bhh0 = (const float*)d_in[4];
  const float* Wih1 = (const float*)d_in[5];
  const float* Whh1 = (const float*)d_in[6];
  const float* bih1 = (const float*)d_in[7];
  const float* bhh1 = (const float*)d_in[8];
  const float* Wlin = (const float*)d_in[9];
  const float* blin = (const float*)d_in[10];
  float* out = (float*)d_out;

  const int B = out_size / 33;       // y(B) + h(16B) + c(16B)
  const int T = in_sizes[0] / B;     // x is [B, T, 1]
  const int waves   = (B + 15) / 16; // 16 batch rows per wave
  const int threads = 256;           // 8 waves per block
  const int blocks  = (waves * 32 + threads - 1) / threads;

  lstm2_wmma_kernel<<<blocks, threads, 0, stream>>>(
      x, Wih0, Whh0, bih0, bhh0, Wih1, Whh1, bih1, bhh1, Wlin, blin,
      out, B, T);
}
